// MultiHeadAttn_48361331753685
// MI455X (gfx1250) — matrix operand order repaired and checked by reading, not compiled, not run on hardware
//
#include <hip/hip_runtime.h>
#include <hip/hip_bf16.h>
#include <cmath>

#define D_MODEL 512
#define N_HEADS 8
#define D_HEAD  64
#define SEQ     2048
#define BATCH   2

typedef _Float16 f16;
typedef __attribute__((ext_vector_type(16))) _Float16 v16h;
typedef __attribute__((ext_vector_type(8)))  _Float16 v8h;
typedef __attribute__((ext_vector_type(4)))  _Float16 v4h;
typedef __attribute__((ext_vector_type(8)))  float    v8f;

__device__ __forceinline__ v8f wmma16(v16h a, v16h b, v8f c) {
  // D = A(16x32 f16) * B(32x16 f16) + C(16x16 f32)
  return __builtin_amdgcn_wmma_f32_16x16x32_f16(false, a, false, b, (short)0, c, false, false);
}

// ---- fragment loaders (wave32; contiguous k order, same for A and B) ----
// A 16x32 f16: lane m = lane&15; element e holds K = e + 16*half (contiguous)
__device__ __forceinline__ v16h load_a_f16(const f16* rk, int half) {
  const v8h* p0 = (const v8h*)(rk + 16 * half);
  const v8h* p1 = (const v8h*)(rk + 8 + 16 * half);
  v8h lo = *p0, hi = *p1;
  v16h a;
#pragma unroll
  for (int i = 0; i < 8; ++i) { a[i] = lo[i]; a[i + 8] = hi[i]; }
  return a;
}
// B 32x16 f16: lane n = lane&15; element e holds K = e + 16*half (contiguous)
__device__ __forceinline__ v16h load_b_f16(const f16* p) {
  const v8h* q = (const v8h*)p;
  v8h lo = q[0], hi = q[1];
  v16h b;
#pragma unroll
  for (int i = 0; i < 8; ++i) { b[i] = lo[i]; b[i + 8] = hi[i]; }
  return b;
}

// ============================================================================
// Kernel 0: one-shot fp32 -> f16 conversion (inputs + weights). Removes all
// v_cvt traffic from the GEMM hot loops and halves their read bytes.
// ============================================================================
__global__ void __launch_bounds__(256)
cvt_f16_kernel(const float* __restrict__ src, f16* __restrict__ dst, int n4) {
  const int i = blockIdx.x * blockDim.x + threadIdx.x;
  if (i < n4) {
    const float4 x = ((const float4*)src)[i];
    v4h o;
    o[0] = (f16)x.x; o[1] = (f16)x.y; o[2] = (f16)x.z; o[3] = (f16)x.w;
    ((v4h*)dst)[i] = o;
  }
}

// ============================================================================
// Kernel 1: Q/K/V projections (all-f16 fragments). out[m,j] = sum_k in[m,k]*W[j,k],
// written per head (j -> d=j>>3, h=j&7) as Qh/Kh [B,H,L,64] f16 and V transposed
// [B,H,64,L]. One wave computes a 16(M) x 64(N) strip for all three weights.
// ============================================================================
__global__ void __launch_bounds__(128)
qkv_proj_kernel(const f16* __restrict__ qc, const f16* __restrict__ kc,
                const f16* __restrict__ vc, const f16* __restrict__ wq,
                const f16* __restrict__ wk, const f16* __restrict__ wv,
                f16* __restrict__ qh, f16* __restrict__ kh, f16* __restrict__ vt) {
  const int lane  = threadIdx.x & 31;
  const int wid   = blockIdx.x * (blockDim.x >> 5) + (threadIdx.x >> 5);
  const int mtile = wid >> 3;   // 0..255  (16 rows each over B*L = 4096)
  const int n0    = (wid & 7) * 64;
  const int row0  = mtile * 16;
  const int nl = lane & 15, half = lane >> 4;
  const int arow = row0 + nl;

  const f16* srcs[3] = {qc, kc, vc};
  const f16* wts[3]  = {wq, wk, wv};
#pragma unroll
  for (int w = 0; w < 3; ++w) {
    const f16* src = srcs[w];
    const f16* W   = wts[w];
    v8f c0 = {}, c1 = {}, c2 = {}, c3 = {};
    for (int k0 = 0; k0 < D_MODEL; k0 += 32) {
      v16h a  = load_a_f16(src + (size_t)arow * D_MODEL + k0, half);
      v16h b0 = load_b_f16(W + (size_t)(n0 +  0 + nl) * D_MODEL + k0 + 16 * half);
      v16h b1 = load_b_f16(W + (size_t)(n0 + 16 + nl) * D_MODEL + k0 + 16 * half);
      v16h b2 = load_b_f16(W + (size_t)(n0 + 32 + nl) * D_MODEL + k0 + 16 * half);
      v16h b3 = load_b_f16(W + (size_t)(n0 + 48 + nl) * D_MODEL + k0 + 16 * half);
      c0 = wmma16(a, b0, c0); c1 = wmma16(a, b1, c1);
      c2 = wmma16(a, b2, c2); c3 = wmma16(a, b3, c3);
    }
    v8f cc[4] = {c0, c1, c2, c3};
#pragma unroll
    for (int t = 0; t < 4; ++t) {
      const int j = n0 + t * 16 + nl;      // C layout: lane holds column j
      const int d = j >> 3, hh = j & 7;    // innermost-head split
      if (w < 2) {
        f16* dst = (w == 0) ? qh : kh;
#pragma unroll
        for (int r = 0; r < 8; ++r) {
          const int rr = row0 + r + 8 * half;          // row index in [0,4096)
          const int b = rr >> 11, l = rr & (SEQ - 1);
          dst[(((size_t)(b * N_HEADS + hh) * SEQ) + l) * D_HEAD + d] = (f16)cc[t][r];
        }
      } else {
        // V^T: consecutive r -> consecutive l, pack two f16 per 32-bit store
#pragma unroll
        for (int r = 0; r < 8; r += 2) {
          const int rr = row0 + r + 8 * half;
          const int b = rr >> 11, l = rr & (SEQ - 1);
          union { f16 h[2]; unsigned u; } pk;
          pk.h[0] = (f16)cc[t][r];
          pk.h[1] = (f16)cc[t][r + 1];
          *(unsigned*)(vt + ((size_t)(b * N_HEADS + hh) * D_HEAD + d) * SEQ + l) = pk.u;
        }
      }
    }
  }
}

// ============================================================================
// Kernel 2: flash attention per (b,h). One wave owns 16 queries, streams all
// 2048 keys in steps of 32. Computes S^T = K*Q^T (lane = query -> lane-local
// softmax stats), builds P^T B-fragment with 8 half-swap shuffles, then
// O^T += V^T * P^T. Writes O as [B,L,512] f16 with col = d*8 + h.
// ============================================================================
__global__ void __launch_bounds__(128)
attn_kernel(const f16* __restrict__ qh, const f16* __restrict__ kh,
            const f16* __restrict__ vt, f16* __restrict__ oh) {
  const int lane  = threadIdx.x & 31;
  const int wid   = blockIdx.x * (blockDim.x >> 5) + (threadIdx.x >> 5);
  const int bh    = wid >> 7;        // 0..15
  const int q0    = (wid & 127) * 16;
  const int b = bh >> 3, h = bh & 7;
  const int nl = lane & 15, half = lane >> 4;

  const f16* Qp = qh + (size_t)(b * N_HEADS + h) * SEQ * D_HEAD;
  const f16* Kp = kh + (size_t)(b * N_HEADS + h) * SEQ * D_HEAD;
  const f16* Vt = vt + (size_t)(b * N_HEADS + h) * D_HEAD * SEQ;

  // Q^T B-fragments (B[k=d][n=query]) held for the whole key sweep
  const v16h qb0 = load_b_f16(Qp + (size_t)(q0 + nl) * D_HEAD +  0 + 16 * half);
  const v16h qb1 = load_b_f16(Qp + (size_t)(q0 + nl) * D_HEAD + 32 + 16 * half);

  v8f o0 = {}, o1 = {}, o2 = {}, o3 = {};   // O^T accumulator (64 d-rows x 16 q)
  float mmax = -INFINITY, lsum = 0.f;
  const float rs = 0.0441941738241592f;     // 1/sqrt(D_MODEL) per reference

  for (int j0 = 0; j0 < SEQ; j0 += 32) {
    // S^T tiles for keys [j0, j0+16) and [j0+16, j0+32)
    v16h ka0 = load_a_f16(Kp + (size_t)(j0 + nl) * D_HEAD + 0, half);
    v16h ka1 = load_a_f16(Kp + (size_t)(j0 + nl) * D_HEAD + 32, half);
    v8f s0 = {};
    s0 = wmma16(ka0, qb0, s0);
    s0 = wmma16(ka1, qb1, s0);
    ka0 = load_a_f16(Kp + (size_t)(j0 + 16 + nl) * D_HEAD + 0, half);
    ka1 = load_a_f16(Kp + (size_t)(j0 + 16 + nl) * D_HEAD + 32, half);
    v8f s1 = {};
    s1 = wmma16(ka0, qb0, s1);
    s1 = wmma16(ka1, qb1, s1);

    if (j0 + 32 < SEQ) {   // gfx1250 global_prefetch_b8 for next key/value tiles
      __builtin_prefetch(Kp + (size_t)(j0 + 32 + nl) * D_HEAD, 0, 1);
      __builtin_prefetch(Vt + (size_t)nl * SEQ + j0 + 32, 0, 1);
    }

    // online softmax; each lane-pair (nl, nl+16) holds the same query
    float tmax = -INFINITY;
#pragma unroll
    for (int r = 0; r < 8; ++r) {
      s0[r] *= rs; s1[r] *= rs;
      tmax = fmaxf(tmax, fmaxf(s0[r], s1[r]));
    }
    tmax = fmaxf(tmax, __shfl_xor(tmax, 16, 32));
    const float nm   = fmaxf(mmax, tmax);
    const float corr = __expf(mmax - nm);
    mmax = nm;
    float ts = 0.f;
#pragma unroll
    for (int r = 0; r < 8; ++r) {
      s0[r] = __expf(s0[r] - nm);
      s1[r] = __expf(s1[r] - nm);
      ts += s0[r] + s1[r];
    }
    ts += __shfl_xor(ts, 16, 32);
    lsum = lsum * corr + ts;

    // P^T B-fragment: element e holds key j0 + e + 16*half (lane = query)
    v16h pb;
#pragma unroll
    for (int r = 0; r < 8; ++r) {
      const float send = half ? s0[r] : s1[r];
      const float recv = __shfl_xor(send, 16, 32);
      pb[r]     = (f16)(half ? recv  : s0[r]);   // keys (0..7)  + 16*half
      pb[r + 8] = (f16)(half ? s1[r] : recv);    // keys (8..15) + 16*half
    }

#pragma unroll
    for (int r = 0; r < 8; ++r) { o0[r] *= corr; o1[r] *= corr; o2[r] *= corr; o3[r] *= corr; }

    const v16h va0 = load_a_f16(Vt + (size_t)( 0 + nl) * SEQ + j0, half);
    const v16h va1 = load_a_f16(Vt + (size_t)(16 + nl) * SEQ + j0, half);
    const v16h va2 = load_a_f16(Vt + (size_t)(32 + nl) * SEQ + j0, half);
    const v16h va3 = load_a_f16(Vt + (size_t)(48 + nl) * SEQ + j0, half);
    o0 = wmma16(va0, pb, o0);
    o1 = wmma16(va1, pb, o1);
    o2 = wmma16(va2, pb, o2);
    o3 = wmma16(va3, pb, o3);
  }

  const float inv = 1.f / lsum;
  const int l = q0 + nl;
  f16* orow = oh + (size_t)(b * SEQ + l) * D_MODEL;
  v8f oo[4] = {o0, o1, o2, o3};
#pragma unroll
  for (int t = 0; t < 4; ++t)
#pragma unroll
    for (int r = 0; r < 8; ++r) {
      const int d = t * 16 + r + 8 * half;
      orow[d * N_HEADS + h] = (f16)(oo[t][r] * inv);
    }
}

// ============================================================================
// Kernel 3: out = O @ Wp^T + q(residual), then LayerNorm (unbiased std) per row.
// One 256-thread block (8 waves) owns 16 rows x all 512 columns so the row
// statistics reduce inside the block (shuffles + tiny LDS).
// ============================================================================
__global__ void __launch_bounds__(256)
proj_ln_kernel(const f16* __restrict__ oh, const f16* __restrict__ wp,
               const float* __restrict__ qres, const float* __restrict__ scale,
               const float* __restrict__ offset, float* __restrict__ out) {
  __shared__ float lsum[16][8];
  __shared__ float lsq[16][8];
  const int lane = threadIdx.x & 31;
  const int w    = threadIdx.x >> 5;   // wave 0..7 -> 64-column strip
  const int n0   = w * 64;
  const int row0 = blockIdx.x * 16;
  const int nl = lane & 15, half = lane >> 4;
  const int arow = row0 + nl;

  v8f c0 = {}, c1 = {}, c2 = {}, c3 = {};
  for (int k0 = 0; k0 < D_MODEL; k0 += 32) {
    v16h a  = load_a_f16(oh + (size_t)arow * D_MODEL + k0, half);
    v16h b0 = load_b_f16(wp + (size_t)(n0 +  0 + nl) * D_MODEL + k0 + 16 * half);
    v16h b1 = load_b_f16(wp + (size_t)(n0 + 16 + nl) * D_MODEL + k0 + 16 * half);
    v16h b2 = load_b_f16(wp + (size_t)(n0 + 32 + nl) * D_MODEL + k0 + 16 * half);
    v16h b3 = load_b_f16(wp + (size_t)(n0 + 48 + nl) * D_MODEL + k0 + 16 * half);
    c0 = wmma16(a, b0, c0); c1 = wmma16(a, b1, c1);
    c2 = wmma16(a, b2, c2); c3 = wmma16(a, b3, c3);
  }
  v8f cc[4] = {c0, c1, c2, c3};

  // residual add (x = proj_out + q), fp32 residual for LN accuracy
#pragma unroll
  for (int t = 0; t < 4; ++t) {
    const int col = n0 + t * 16 + nl;
#pragma unroll
    for (int r = 0; r < 8; ++r) {
      const int rr = row0 + r + 8 * half;
      cc[t][r] += qres[(size_t)rr * D_MODEL + col];
    }
  }

  // per-row mean / unbiased std over 512 columns
#pragma unroll
  for (int r = 0; r < 8; ++r) {
    float s = 0.f, s2 = 0.f;
#pragma unroll
    for (int t = 0; t < 4; ++t) { const float x = cc[t][r]; s += x; s2 += x * x; }
#pragma unroll
    for (int m = 1; m < 16; m <<= 1) { s += __shfl_xor(s, m, 32); s2 += __shfl_xor(s2, m, 32); }
    if (nl == 0) { lsum[r + 8 * half][w] = s; lsq[r + 8 * half][w] = s2; }
  }
  __syncthreads();

  float mean_[8], dn_[8];
#pragma unroll
  for (int r = 0; r < 8; ++r) {
    const int rid = r + 8 * half;
    float s = 0.f, s2 = 0.f;
#pragma unroll
    for (int ww = 0; ww < 8; ++ww) { s += lsum[rid][ww]; s2 += lsq[rid][ww]; }
    const float mean = s * (1.f / 512.f);
    float var = (s2 - 512.f * mean * mean) * (1.f / 511.f);
    var = fmaxf(var, 0.f);
    mean_[r] = mean;
    dn_[r]   = 1.f / (sqrtf(var) + 1e-9f);
  }

#pragma unroll
  for (int t = 0; t < 4; ++t) {
    const int col = n0 + t * 16 + nl;
    const float sc = scale[col], of = offset[col];
#pragma unroll
    for (int r = 0; r < 8; ++r) {
      const int rr = row0 + r + 8 * half;
      out[(size_t)rr * D_MODEL + col] = sc * (cc[t][r] - mean_[r]) * dn_[r] + of;
    }
  }
}

extern "C" void kernel_launch(void* const* d_in, const int* in_sizes, int n_in,
                              void* d_out, int out_size, void* d_ws, size_t ws_size,
                              hipStream_t stream) {
  (void)in_sizes; (void)n_in; (void)out_size; (void)ws_size;
  const float* q      = (const float*)d_in[0];
  const float* k      = (const float*)d_in[1];
  const float* v      = (const float*)d_in[2];
  const float* Wq     = (const float*)d_in[3];
  const float* Wk     = (const float*)d_in[4];
  const float* Wv     = (const float*)d_in[5];
  const float* Wp     = (const float*)d_in[6];
  const float* scale  = (const float*)d_in[7];
  const float* offset = (const float*)d_in[8];
  float* out = (float*)d_out;

  char* ws = (char*)d_ws;
  const size_t HSZ = (size_t)BATCH * N_HEADS * SEQ * D_HEAD * sizeof(f16); // 4 MB
  const size_t ISZ = (size_t)BATCH * SEQ * D_MODEL * sizeof(f16);          // 4 MB
  const size_t WSZ = (size_t)D_MODEL * D_MODEL * sizeof(f16);              // 512 KB
  size_t off = 0;
  f16* qh  = (f16*)(ws + off); off += HSZ;   // [B,H,L,64]
  f16* kh  = (f16*)(ws + off); off += HSZ;   // [B,H,L,64]
  f16* vt  = (f16*)(ws + off); off += HSZ;   // [B,H,64,L]  (V transposed per head)
  f16* oh  = (f16*)(ws + off); off += HSZ;   // [B,L,512]   (col = d*8 + h)
  f16* qc  = (f16*)(ws + off); off += ISZ;   // f16 copies of the inputs
  f16* kc  = (f16*)(ws + off); off += ISZ;
  f16* vc  = (f16*)(ws + off); off += ISZ;
  f16* wq16 = (f16*)(ws + off); off += WSZ;  // f16 copies of the weights
  f16* wk16 = (f16*)(ws + off); off += WSZ;
  f16* wv16 = (f16*)(ws + off); off += WSZ;
  f16* wp16 = (f16*)(ws + off); off += WSZ;

  const int nI4 = BATCH * SEQ * D_MODEL / 4;      // 524288
  const int nW4 = D_MODEL * D_MODEL / 4;          // 65536
  cvt_f16_kernel<<<(nI4 + 255) / 256, 256, 0, stream>>>(q,  qc,   nI4);
  cvt_f16_kernel<<<(nI4 + 255) / 256, 256, 0, stream>>>(k,  kc,   nI4);
  cvt_f16_kernel<<<(nI4 + 255) / 256, 256, 0, stream>>>(v,  vc,   nI4);
  cvt_f16_kernel<<<(nW4 + 255) / 256, 256, 0, stream>>>(Wq, wq16, nW4);
  cvt_f16_kernel<<<(nW4 + 255) / 256, 256, 0, stream>>>(Wk, wk16, nW4);
  cvt_f16_kernel<<<(nW4 + 255) / 256, 256, 0, stream>>>(Wv, wv16, nW4);
  cvt_f16_kernel<<<(nW4 + 255) / 256, 256, 0, stream>>>(Wp, wp16, nW4);

  // 2048 waves each: projections (16x64 strips x3), attention (16 queries),
  // then 256 blocks for the fused epilogue.
  qkv_proj_kernel<<<512, 128, 0, stream>>>(qc, kc, vc, wq16, wk16, wv16, qh, kh, vt);
  attn_kernel   <<<512, 128, 0, stream>>>(qh, kh, vt, oh);
  proj_ln_kernel<<<256, 256, 0, stream>>>(oh, wp16, q, scale, offset, out);
}
